// Seq2SeqModel_80504866996965
// MI455X (gfx1250) — compile-verified, matrix operand order checked by reading, not run on hardware
//
#include <hip/hip_runtime.h>
#include <hip/hip_bf16.h>

// ---------------------------------------------------------------------------
// Seq2Seq LSTM (V=32000,E=128,H=256,B=32,S=T=128) for gfx1250 (CDNA5, wave32)
//  - All GEMMs via v_wmma_f32_16x16x32_bf16 (fp32 accumulate).
//  - Recurrent scan: 8 cooperating workgroups (one per WGP), each with its
//    64KB Whh slice resident in LDS; gate slices exchanged through an L2
//    ping-pong buffer with one grid barrier per step; h/c updated redundantly
//    in each WGP's LDS (no second barrier).
//  - Logits projection [4096x256]x[256x32000]: block stages the 32KB W_out
//    strip in LDS (read once per block); each wave computes 32x64 with A
//    fragments reused x4 and B fragments served from LDS. Balances L2 traffic
//    against the 524MB HBM output stream.
// ---------------------------------------------------------------------------

typedef __attribute__((ext_vector_type(16))) __bf16 bf16x16;
typedef __attribute__((ext_vector_type(8)))  __bf16 bf16x8;
typedef __attribute__((ext_vector_type(8)))  float  f32x8;

namespace {
constexpr int kV = 32000, kE = 128, kH = 256, kH4 = 1024;
constexpr int kB = 32, kS = 128, kT = 128;
constexpr int kMrows = kB * kS;            // 4096 (same for enc & dec)
constexpr int kP = 8;                      // workgroups cooperating on the scan
constexpr int kNslice = kH4 / kP;          // 128 gate columns per workgroup
}

// 16x32 bf16 fragment: 16-wide dim indexes rows r0..r0+15 of a row-major
// [rows x ld] array, K contiguous. Matches CDNA5 16-bit A/B VGPR layout:
//   lanes 0-15 : elems 0..7 = K k0..k0+7,  elems 8..15 = K k0+16..k0+23
//   lanes 16-31: elems 0..7 = K k0+8..+15, elems 8..15 = K k0+24..+31
static __device__ __forceinline__
bf16x16 load_frag(const __bf16* __restrict__ base, int ld, int r0, int k0) {
  const int lane = threadIdx.x & 31;
  const int half = lane >> 4;
  const __bf16* p = base + (size_t)(r0 + (lane & 15)) * ld + k0 + half * 8;
  bf16x8 lo = *reinterpret_cast<const bf16x8*>(p);
  bf16x8 hi = *reinterpret_cast<const bf16x8*>(p + 16);
  bf16x16 v;
#pragma unroll
  for (int i = 0; i < 8; ++i) { v[i] = lo[i]; v[8 + i] = hi[i]; }
  return v;
}

// ---------------------------------------------------------------------------
__global__ void f32_to_bf16_kernel(const float* __restrict__ src,
                                   __bf16* __restrict__ dst, int n) {
  int i = blockIdx.x * blockDim.x + threadIdx.x;
  if (i < n) dst[i] = (__bf16)src[i];
}

__global__ void init_sync_kernel(int* sync) {
  if (threadIdx.x < 8) sync[threadIdx.x] = 0;
}

// Gather token embeddings and convert to bf16
__global__ void embed_bf16_kernel(const int* __restrict__ tok,
                                  const float* __restrict__ emb,
                                  __bf16* __restrict__ out, int ntok) {
  int i = blockIdx.x * blockDim.x + threadIdx.x;
  if (i >= ntok * kE) return;
  int t = i >> 7;
  int e = i & (kE - 1);
  out[i] = (__bf16)emb[(size_t)tok[t] * kE + e];
}

// xg[M x 4H] = A[M x E] * Wih^T + (bih + bhh); B[k,n] = Wih[n][k]
// One wave -> 16x64 strip (4 N-tiles, A fragment reused).
__global__ void xg_gemm_kernel(const __bf16* __restrict__ A,
                               const __bf16* __restrict__ Wih,   // [4H x E] bf16
                               const float* __restrict__ bih,
                               const float* __restrict__ bhh,
                               float* __restrict__ xg, int Mrows) {
  const int wave = (blockIdx.x * blockDim.x + threadIdx.x) >> 5;
  const int ntg = kH4 / 64;                 // 16 groups of 4 N-tiles
  const int mt = wave / ntg, ng = wave % ntg;
  if (mt * 16 >= Mrows) return;
  const int m0 = mt * 16, n0 = ng * 64;

  f32x8 acc[4] = {f32x8{}, f32x8{}, f32x8{}, f32x8{}};
#pragma unroll
  for (int k0 = 0; k0 < kE; k0 += 32) {
    bf16x16 a = load_frag(A, kE, m0, k0);
#pragma unroll
    for (int j = 0; j < 4; ++j) {
      bf16x16 b = load_frag(Wih, kE, n0 + 16 * j, k0);
      acc[j] = __builtin_amdgcn_wmma_f32_16x16x32_bf16(false, a, false, b,
                                                       (short)0, acc[j], false, false);
    }
  }
  const int lane = threadIdx.x & 31, half = lane >> 4;
#pragma unroll
  for (int j = 0; j < 4; ++j) {
    const int n = n0 + 16 * j + (lane & 15);
    const float bias = bih[n] + bhh[n];
#pragma unroll
    for (int v = 0; v < 8; ++v) {
      int m = m0 + v + half * 8;
      xg[(size_t)m * kH4 + n] = acc[j][v] + bias;
    }
  }
}

// ---------------------------------------------------------------------------
// Grid barrier across kP blocks (counter + monotonically increasing flag).
static __device__ __forceinline__
void grid_barrier(int* cnt, int* flag, int* local_phase) {
  __threadfence();                 // make this wave's gate stores visible (agent)
  __syncthreads();
  if (threadIdx.x == 0) {
    ++(*local_phase);
    int t = __hip_atomic_fetch_add(cnt, 1, __ATOMIC_ACQ_REL, __HIP_MEMORY_SCOPE_AGENT);
    if (t == kP - 1) {
      __hip_atomic_store(cnt, 0, __ATOMIC_RELAXED, __HIP_MEMORY_SCOPE_AGENT);
      __hip_atomic_store(flag, *local_phase, __ATOMIC_RELEASE, __HIP_MEMORY_SCOPE_AGENT);
    } else {
      while (__hip_atomic_load(flag, __ATOMIC_ACQUIRE, __HIP_MEMORY_SCOPE_AGENT) <
             *local_phase) {
        __builtin_amdgcn_s_sleep(1);
      }
    }
  }
  __syncthreads();
}

// Cooperative LSTM scan: kP blocks x 512 threads (16 wave32 waves each).
// Block wg owns gate columns [wg*128, wg*128+128); its Whh slice (64KB bf16)
// lives in LDS for the whole scan. h (bf16) and c (f32) are replicated in
// every block's LDS and updated redundantly each step from the shared gates.
__global__ void __launch_bounds__(512, 1)
lstm_seq_kernel(const float* __restrict__ xg,        // [B x Steps x 4H]
                const __bf16* __restrict__ Whh,      // [4H x H] bf16
                const float* __restrict__ h0,
                const float* __restrict__ c0,
                float* __restrict__ hT, float* __restrict__ cT,
                __bf16* __restrict__ out_h,          // [B x Steps x H] (if write_out)
                float* __restrict__ gbuf0,           // [B x 4H] ping
                float* __restrict__ gbuf1,           // [B x 4H] pong
                int* __restrict__ sync,              // {cnt, flag}
                int Steps, int zero_init, int write_out) {
  extern __shared__ char smem_raw[];
  __bf16* whh_s = reinterpret_cast<__bf16*>(smem_raw);                       // 64KB
  float*  c_s   = reinterpret_cast<float*>(smem_raw + kNslice * kH * 2);     // 32KB
  __bf16* h_s   = reinterpret_cast<__bf16*>(smem_raw + kNslice * kH * 2
                                                     + kB * kH * 4);         // 16KB
  const int tid  = threadIdx.x;
  const int lane = tid & 31;
  const int wave = tid >> 5;              // 0..15
  const int half = lane >> 4;
  const int wg   = blockIdx.x;            // 0..kP-1
  const int n_base = wg * kNslice;

  // Preload this block's Whh slice (rows n_base..n_base+127, all 256 K) -> LDS
  {
    const uint4* src = reinterpret_cast<const uint4*>(Whh + (size_t)n_base * kH);
    uint4* dst = reinterpret_cast<uint4*>(whh_s);
    const int nvec = kNslice * kH * 2 / 16;            // 4096
    for (int i = tid; i < nvec; i += 512) dst[i] = src[i];
  }
  // Init replicated state
  for (int e = tid; e < kB * kH; e += 512) {
    float hv = zero_init ? 0.f : h0[e];
    float cv = zero_init ? 0.f : c0[e];
    h_s[e] = (__bf16)hv;
    c_s[e] = cv;
  }
  __syncthreads();

  // Fixed tile per wave: 2 M-tiles x 8 local N-tiles = 16 tiles = 16 waves
  const int mt = wave & 1, nt = wave >> 1;
  const int m0 = mt * 16;
  const int nloc0 = nt * 16;
  const int ncol = n_base + nloc0 + (lane & 15);       // global gate column

  int phase = 0;
  for (int t = 0; t < Steps; ++t) {
    float* gw = (t & 1) ? gbuf1 : gbuf0;

    // ---- GEMM: gates slice = xg[:,t,cols] + h @ Whh_slice^T
    f32x8 acc;
#pragma unroll
    for (int v = 0; v < 8; ++v) {
      int m = m0 + v + half * 8;
      acc[v] = xg[((size_t)m * Steps + t) * kH4 + ncol];
    }
#pragma unroll
    for (int k0 = 0; k0 < kH; k0 += 32) {
      bf16x16 a = load_frag(h_s, kH, m0, k0);          // LDS
      bf16x16 b = load_frag(whh_s, kH, nloc0, k0);     // LDS
      acc = __builtin_amdgcn_wmma_f32_16x16x32_bf16(false, a, false, b,
                                                    (short)0, acc, false, false);
    }
#pragma unroll
    for (int v = 0; v < 8; ++v) {                      // publish (agent scope)
      int m = m0 + v + half * 8;
      __hip_atomic_store(&gw[m * kH4 + ncol], acc[v],
                         __ATOMIC_RELAXED, __HIP_MEMORY_SCOPE_AGENT);
    }

    grid_barrier(&sync[0], &sync[1], &phase);

    // ---- Redundant pointwise update of replicated h/c (16 elems/thread)
#pragma unroll
    for (int r = 0; r < 16; ++r) {
      const int e = tid + r * 512;
      const int b = e >> 8;
      const int j = e & (kH - 1);
      float gi = __hip_atomic_load(&gw[b * kH4 + j],          __ATOMIC_RELAXED, __HIP_MEMORY_SCOPE_AGENT);
      float gf = __hip_atomic_load(&gw[b * kH4 + kH + j],     __ATOMIC_RELAXED, __HIP_MEMORY_SCOPE_AGENT);
      float gg = __hip_atomic_load(&gw[b * kH4 + 2 * kH + j], __ATOMIC_RELAXED, __HIP_MEMORY_SCOPE_AGENT);
      float go = __hip_atomic_load(&gw[b * kH4 + 3 * kH + j], __ATOMIC_RELAXED, __HIP_MEMORY_SCOPE_AGENT);
      float i_ = 1.f / (1.f + __expf(-gi));
      float f_ = 1.f / (1.f + __expf(-gf));
      float g_ = tanhf(gg);
      float o_ = 1.f / (1.f + __expf(-go));
      float cn = f_ * c_s[e] + i_ * g_;
      float hn = o_ * tanhf(cn);
      c_s[e] = cn;
      h_s[e] = (__bf16)hn;
      if (write_out && wg == 0)
        out_h[((size_t)b * Steps + t) * kH + j] = (__bf16)hn;
    }
    __syncthreads();   // h_s ready for next step's GEMM (cross-WG via ping-pong)
  }

  if (wg == 0) {
    for (int e = tid; e < kB * kH; e += 512) {
      hT[e] = (float)h_s[e];
      cT[e] = c_s[e];
    }
  }
}

// logits[M x V] = dec_out[M x H] * W_out^T + b_out
// Block (8 waves): 256x64 tile. The 64-column W_out strip (64x256 bf16 = 32KB)
// is staged in LDS once per block; wave w computes rows [w*32, w*32+32).
__global__ void __launch_bounds__(256, 1)
logits_gemm_kernel(const __bf16* __restrict__ A,     // [4096 x 256]
                   const __bf16* __restrict__ Wout,  // [V x H] bf16
                   const float* __restrict__ b_out,
                   float* __restrict__ logits) {
  __shared__ __bf16 b_s[64 * kH];                    // 32KB strip
  const int mg = blockIdx.x / (kV / 64);             // 0..15  (256-row group)
  const int ng = blockIdx.x % (kV / 64);             // 0..499 (64-col strip)
  const int n0 = ng * 64;

  // Stage W_out strip rows n0..n0+63 into LDS (2048 x uint4)
  {
    const uint4* src = reinterpret_cast<const uint4*>(Wout + (size_t)n0 * kH);
    uint4* dst = reinterpret_cast<uint4*>(b_s);
#pragma unroll
    for (int i = 0; i < 8; ++i) dst[threadIdx.x + i * 256] = src[threadIdx.x + i * 256];
  }
  __syncthreads();

  const int wave = threadIdx.x >> 5;                 // 0..7
  const int m0 = mg * 256 + wave * 32;               // 32 rows per wave

  f32x8 acc[2][4] = {{f32x8{}, f32x8{}, f32x8{}, f32x8{}},
                     {f32x8{}, f32x8{}, f32x8{}, f32x8{}}};
#pragma unroll
  for (int k0 = 0; k0 < kH; k0 += 32) {
    if (k0 + 32 < kH) {
      // hint the next A rows into cache (global_prefetch)
      __builtin_prefetch(A + (size_t)m0 * kH + k0 + 32, 0, 1);
      __builtin_prefetch(A + (size_t)(m0 + 16) * kH + k0 + 32, 0, 1);
    }
    bf16x16 a0 = load_frag(A, kH, m0, k0);
    bf16x16 a1 = load_frag(A, kH, m0 + 16, k0);
#pragma unroll
    for (int j = 0; j < 4; ++j) {
      bf16x16 b = load_frag(b_s, kH, 16 * j, k0);    // LDS
      acc[0][j] = __builtin_amdgcn_wmma_f32_16x16x32_bf16(false, a0, false, b,
                                                          (short)0, acc[0][j], false, false);
      acc[1][j] = __builtin_amdgcn_wmma_f32_16x16x32_bf16(false, a1, false, b,
                                                          (short)0, acc[1][j], false, false);
    }
  }
  const int lane = threadIdx.x & 31, half = lane >> 4;
#pragma unroll
  for (int j = 0; j < 4; ++j) {
    const int n = n0 + 16 * j + (lane & 15);
    const float bias = b_out[n];
#pragma unroll
    for (int i = 0; i < 2; ++i) {
#pragma unroll
      for (int v = 0; v < 8; ++v) {
        int m = m0 + 16 * i + v + half * 8;
        logits[(size_t)m * kV + n] = acc[i][j][v] + bias;
      }
    }
  }
}

// ---------------------------------------------------------------------------
extern "C" void kernel_launch(void* const* d_in, const int* in_sizes, int n_in,
                              void* d_out, int out_size, void* d_ws, size_t ws_size,
                              hipStream_t stream) {
  (void)in_sizes; (void)n_in; (void)out_size; (void)ws_size;

  const int*   x       = (const int*)  d_in[0];
  const int*   y       = (const int*)  d_in[1];
  const float* enc_emb = (const float*)d_in[2];
  const float* dec_emb = (const float*)d_in[3];
  const float* Wih_enc = (const float*)d_in[4];
  const float* Whh_enc = (const float*)d_in[5];
  const float* bih_enc = (const float*)d_in[6];
  const float* bhh_enc = (const float*)d_in[7];
  const float* Wih_dec = (const float*)d_in[8];
  const float* Whh_dec = (const float*)d_in[9];
  const float* bih_dec = (const float*)d_in[10];
  const float* bhh_dec = (const float*)d_in[11];
  const float* W_out   = (const float*)d_in[12];
  const float* b_out   = (const float*)d_in[13];
  float* logits = (float*)d_out;

  char* ws = (char*)d_ws;
  size_t off = 0;
  auto carve = [&](size_t bytes) -> char* {
    char* p = ws + off;
    off += (bytes + 255) & ~(size_t)255;
    return p;
  };
  __bf16* pWihE = (__bf16*)carve((size_t)kH4 * kE * 2);
  __bf16* pWhhE = (__bf16*)carve((size_t)kH4 * kH * 2);
  __bf16* pWihD = (__bf16*)carve((size_t)kH4 * kE * 2);
  __bf16* pWhhD = (__bf16*)carve((size_t)kH4 * kH * 2);
  __bf16* pWout = (__bf16*)carve((size_t)kV  * kH * 2);
  __bf16* pxemb = (__bf16*)carve((size_t)kB * kS * kE * 2);
  __bf16* pyemb = (__bf16*)carve((size_t)kB * kT * kE * 2);
  float*  pxgE  = (float*) carve((size_t)kB * kS * kH4 * 4);
  float*  pxgD  = (float*) carve((size_t)kB * kT * kH4 * 4);
  float*  phT_e = (float*) carve((size_t)kB * kH * 4);
  float*  pcT_e = (float*) carve((size_t)kB * kH * 4);
  float*  phT_d = (float*) carve((size_t)kB * kH * 4);
  float*  pcT_d = (float*) carve((size_t)kB * kH * 4);
  __bf16* pdec  = (__bf16*)carve((size_t)kB * kT * kH * 2);
  float*  pgb0  = (float*) carve((size_t)kB * kH4 * 4);
  float*  pgb1  = (float*) carve((size_t)kB * kH4 * 4);
  int*    psync = (int*)   carve(256);

  const dim3 blk(256);

  // 0) Reset grid-barrier state (fresh every call; graph-replay safe)
  init_sync_kernel<<<1, 32, 0, stream>>>(psync);

  // 1) Weights -> bf16
  f32_to_bf16_kernel<<<(kH4 * kE + 255) / 256, blk, 0, stream>>>(Wih_enc, pWihE, kH4 * kE);
  f32_to_bf16_kernel<<<(kH4 * kH + 255) / 256, blk, 0, stream>>>(Whh_enc, pWhhE, kH4 * kH);
  f32_to_bf16_kernel<<<(kH4 * kE + 255) / 256, blk, 0, stream>>>(Wih_dec, pWihD, kH4 * kE);
  f32_to_bf16_kernel<<<(kH4 * kH + 255) / 256, blk, 0, stream>>>(Whh_dec, pWhhD, kH4 * kH);
  f32_to_bf16_kernel<<<(kV * kH + 255) / 256, blk, 0, stream>>>(W_out, pWout, kV * kH);

  // 2) Embedding gather (+bf16)
  embed_bf16_kernel<<<(kB * kS * kE) / 256, blk, 0, stream>>>(x, enc_emb, pxemb, kB * kS);
  embed_bf16_kernel<<<(kB * kT * kE) / 256, blk, 0, stream>>>(y, dec_emb, pyemb, kB * kT);

  // 3) Input-gate GEMMs: xg = emb @ Wih^T + (bih + bhh)
  {
    const int waves = (kMrows / 16) * (kH4 / 64);     // 4096
    xg_gemm_kernel<<<waves / 8, blk, 0, stream>>>(pxemb, pWihE, bih_enc, bhh_enc, pxgE, kMrows);
    xg_gemm_kernel<<<waves / 8, blk, 0, stream>>>(pyemb, pWihD, bih_dec, bhh_dec, pxgD, kMrows);
  }

  // 4) Cooperative recurrent scans: 8 WGPs, Whh slices LDS-resident
  const size_t lstm_smem = (size_t)kNslice * kH * 2   // Whh slice  64KB
                         + (size_t)kB * kH * 4        // c          32KB
                         + (size_t)kB * kH * 2;       // h          16KB
  lstm_seq_kernel<<<kP, 512, lstm_smem, stream>>>(pxgE, pWhhE, nullptr, nullptr,
                                                  phT_e, pcT_e, nullptr,
                                                  pgb0, pgb1, psync, kS,
                                                  /*zero_init=*/1, /*write_out=*/0);
  lstm_seq_kernel<<<kP, 512, lstm_smem, stream>>>(pxgD, pWhhD, phT_e, pcT_e,
                                                  phT_d, pcT_d, pdec,
                                                  pgb0, pgb1, psync, kT,
                                                  /*zero_init=*/0, /*write_out=*/1);

  // 5) Output projection: logits = dec_out @ W_out^T + b_out
  //    16 row-groups x 500 col-strips = 8000 blocks of 256 threads
  logits_gemm_kernel<<<16 * (kV / 64), blk, 0, stream>>>(pdec, pWout, b_out, logits);
}
